// LearnableWatershed_22119081574494
// MI455X (gfx1250) — matrix operand order repaired and checked by reading, not compiled, not run on hardware
//
#include <hip/hip_runtime.h>
#include <hip/hip_bf16.h>
#include <math.h>

typedef __attribute__((ext_vector_type(16))) _Float16 v16h;
typedef __attribute__((ext_vector_type(8)))  _Float16 v8h;
typedef __attribute__((ext_vector_type(8)))  float    v8f;

#define BATCH 8
#define HH 512
#define WW 512
#define HWSZ (HH * WW)
#define NMARK 32
#define NITER 10

// ---------------------------------------------------------------------------
// Stage 1: gray = 0.2989 r + 0.587 g + 0.114 b ; gradient = hypot(Sobel)
// ---------------------------------------------------------------------------
__global__ void lw_gray_grad(const float* __restrict__ img,
                             const float* __restrict__ Kx,
                             const float* __restrict__ Ky,
                             float* __restrict__ gray,
                             float* __restrict__ grad) {
    long long p = (long long)blockIdx.x * blockDim.x + threadIdx.x;
    if (p >= (long long)BATCH * HWSZ) return;
    int b = (int)(p / HWSZ);
    int rem = (int)(p % HWSZ);
    int y = rem / WW, x = rem % WW;
    const float* rC = img + (long long)(b * 3 + 0) * HWSZ;
    const float* gC = img + (long long)(b * 3 + 1) * HWSZ;
    const float* bC = img + (long long)(b * 3 + 2) * HWSZ;

    float gx = 0.f, gy = 0.f, gcenter = 0.f;
    #pragma unroll
    for (int ky = 0; ky < 3; ++ky) {
        #pragma unroll
        for (int kx = 0; kx < 3; ++kx) {
            int yy = y + ky - 1, xx = x + kx - 1;
            float gv = 0.f;
            if (yy >= 0 && yy < HH && xx >= 0 && xx < WW) {
                int q = yy * WW + xx;
                gv = 0.2989f * rC[q] + 0.587f * gC[q] + 0.114f * bC[q];
            }
            gx += Kx[ky * 3 + kx] * gv;
            gy += Ky[ky * 3 + kx] * gv;
            if (ky == 1 && kx == 1) gcenter = gv;
        }
    }
    gray[p] = gcenter;
    grad[p] = sqrtf(gx * gx + gy * gy);
}

// ---------------------------------------------------------------------------
// Stage 2: conv1 (1 -> 16 channels, 3x3, bias, relu), direct (K=9 too small
// for WMMA to pay off). Output fp16 NCHW.
// ---------------------------------------------------------------------------
__global__ void lw_conv1(const float* __restrict__ gray,
                         const float* __restrict__ w1,
                         const float* __restrict__ b1,
                         _Float16* __restrict__ x1) {
    long long p = (long long)blockIdx.x * blockDim.x + threadIdx.x;
    if (p >= (long long)BATCH * HWSZ) return;
    int b = (int)(p / HWSZ);
    int rem = (int)(p % HWSZ);
    int y = rem / WW, x = rem % WW;

    float patch[9];
    #pragma unroll
    for (int t = 0; t < 9; ++t) {
        int yy = y + t / 3 - 1, xx = x + t % 3 - 1;
        patch[t] = (yy >= 0 && yy < HH && xx >= 0 && xx < WW)
                       ? gray[(long long)b * HWSZ + yy * WW + xx] : 0.f;
    }
    #pragma unroll
    for (int co = 0; co < 16; ++co) {
        float acc = b1[co];
        #pragma unroll
        for (int t = 0; t < 9; ++t) acc = fmaf(w1[co * 9 + t], patch[t], acc);
        acc = fmaxf(acc, 0.f);
        x1[((long long)(b * 16 + co)) * HWSZ + rem] = (_Float16)acc;
    }
}

// ---------------------------------------------------------------------------
// Stages 3/4: implicit-GEMM conv via WMMA f16 -> f32.
//   D(32 x 16) = W(32 x K) * Patch(K x 16),  K = Cin*9 padded to mult of 32.
// One wave owns a 32-channel x 16-pixel tile; 2 waves per block.
// A/B fragments built from LDS in contiguous k order (k = 16*hi16 + s), same for A and B:
//   A (16x32, MxK): lane<16 row=lane holds K 0..15, lanes>=16 hold K 16..31.
//   B (32x16, KxN): lane<16 col=lane holds K 0..15, lanes>=16 hold K 16..31.
// ---------------------------------------------------------------------------
template <int CIN, bool RELU>
__global__ void lw_conv_wmma(const _Float16* __restrict__ in,
                             const float* __restrict__ w,
                             const float* __restrict__ bias,
                             _Float16* __restrict__ out) {
    constexpr int KTOT = CIN * 9;
    constexpr int KPAD = ((KTOT + 31) / 32) * 32;
    constexpr int TILES_X = WW / 16;

    __shared__ _Float16 wlds[32 * KPAD];        // weights: [co][k]
    __shared__ _Float16 plds[2 * 16 * KPAD];    // per-wave im2col: [n][k]

    const int tid = threadIdx.x;
    const int lane = tid & 31;
    const int wave = tid >> 5;

    // ---- stage weights (whole block) ----
    for (int idx = tid; idx < 32 * KPAD; idx += 64) {
        int co = idx / KPAD, k = idx % KPAD;
        wlds[idx] = (k < KTOT) ? (_Float16)w[co * KTOT + k] : (_Float16)0.0f;
    }

    // ---- tile coordinates for this wave ----
    const int g  = blockIdx.x * 2 + wave;
    const int xt = g & (TILES_X - 1);
    const int y  = (g >> 5) & (HH - 1);
    const int b  = g >> 14;           // / (TILES_X * HH)
    const int x0 = xt * 16;

    // ---- stage im2col patch for 16 pixels (per wave) ----
    _Float16* pl = plds + wave * 16 * KPAD;
    for (int idx = lane; idx < 16 * KPAD; idx += 32) {
        int n = idx / KPAD, k = idx % KPAD;
        _Float16 v = (_Float16)0.0f;
        if (k < KTOT) {
            int ci = k / 9, t = k % 9;
            int yy = y + t / 3 - 1;
            int xx = x0 + n + t % 3 - 1;
            if (yy >= 0 && yy < HH && xx >= 0 && xx < WW)
                v = in[((long long)(b * CIN + ci)) * HWSZ + yy * WW + xx];
        }
        pl[idx] = v;
    }
    __syncthreads();

    const int mrow = lane & 15;
    const int hi16 = (lane >> 4);          // 0 for lanes 0-15, 1 for 16-31

    v8f c0 = {};
    v8f c1 = {};

    #pragma unroll
    for (int kb = 0; kb < KPAD / 32; ++kb) {
        const int kbase = kb * 32;
        // B fragment: 16 contiguous K values (32B aligned since KPAD%32==0)
        v16h bf = *(const v16h*)&pl[mrow * KPAD + kbase + hi16 * 16];
        // A fragments: two aligned 8-half chunks per tile
        int ka = kbase + hi16 * 16;
        v8h a0lo = *(const v8h*)&wlds[mrow * KPAD + ka];
        v8h a0hi = *(const v8h*)&wlds[mrow * KPAD + ka + 8];
        v8h a1lo = *(const v8h*)&wlds[(mrow + 16) * KPAD + ka];
        v8h a1hi = *(const v8h*)&wlds[(mrow + 16) * KPAD + ka + 8];
        v16h a0, a1;
        #pragma unroll
        for (int i = 0; i < 8; ++i) {
            a0[i] = a0lo[i]; a0[i + 8] = a0hi[i];
            a1[i] = a1lo[i]; a1[i + 8] = a1hi[i];
        }
        c0 = __builtin_amdgcn_wmma_f32_16x16x32_f16(false, a0, false, bf,
                                                    (short)0, c0, false, false);
        c1 = __builtin_amdgcn_wmma_f32_16x16x32_f16(false, a1, false, bf,
                                                    (short)0, c1, false, false);
    }

    // ---- bias (+relu) and store; C/D layout: VGPR r -> M = r + hi16*8 ----
    const int n = mrow;
    #pragma unroll
    for (int r = 0; r < 8; ++r) {
        int co0 = r + hi16 * 8;
        float v0 = c0[r] + bias[co0];
        float v1 = c1[r] + bias[co0 + 16];
        if (RELU) { v0 = fmaxf(v0, 0.f); v1 = fmaxf(v1, 0.f); }
        long long base = (long long)b * 32 * HWSZ + (long long)y * WW + x0 + n;
        out[base + (long long)co0 * HWSZ]        = (_Float16)v0;
        out[base + (long long)(co0 + 16) * HWSZ] = (_Float16)v1;
    }
}

// ---------------------------------------------------------------------------
// Stage 5: markers = argmax_c(logits) (softmax is monotone); one-hot init.
// ---------------------------------------------------------------------------
__global__ void lw_argmax_onehot(const _Float16* __restrict__ logits,
                                 _Float16* __restrict__ lab) {
    long long p = (long long)blockIdx.x * blockDim.x + threadIdx.x;
    if (p >= (long long)BATCH * HWSZ) return;
    int b = (int)(p / HWSZ);
    int rem = (int)(p % HWSZ);
    long long base = (long long)b * NMARK * HWSZ + rem;
    float best = -1e30f; int bi = 0;
    #pragma unroll
    for (int c = 0; c < NMARK; ++c) {
        float v = (float)logits[base + (long long)c * HWSZ];
        if (v > best) { best = v; bi = c; }
    }
    #pragma unroll
    for (int c = 0; c < NMARK; ++c)
        lab[base + (long long)c * HWSZ] = (c == bi) ? (_Float16)1.0f : (_Float16)0.0f;
}

// ---------------------------------------------------------------------------
// Stage 6 (x10): affinity = avgpool3(lab); lab' = softmax_c(affinity + grad)
// ---------------------------------------------------------------------------
__global__ void lw_step(const _Float16* __restrict__ src,
                        const float* __restrict__ grad,
                        _Float16* __restrict__ dst) {
    long long p = (long long)blockIdx.x * blockDim.x + threadIdx.x;
    if (p >= (long long)BATCH * HWSZ) return;
    int b = (int)(p / HWSZ);
    int rem = (int)(p % HWSZ);
    int y = rem / WW, x = rem % WW;
    const float gval = grad[p];
    long long base = (long long)b * NMARK * HWSZ;

    float vals[NMARK];
    float vmax = -1e30f;
    #pragma unroll 4
    for (int c = 0; c < NMARK; ++c) {
        const _Float16* ch = src + base + (long long)c * HWSZ;
        __builtin_prefetch(ch + rem, 0, 1);      // global_prefetch_b8
        float s = 0.f;
        #pragma unroll
        for (int ky = 0; ky < 3; ++ky) {
            int yy = y + ky - 1;
            if (yy < 0 || yy >= HH) continue;
            #pragma unroll
            for (int kx = 0; kx < 3; ++kx) {
                int xx = x + kx - 1;
                if (xx < 0 || xx >= WW) continue;
                s += (float)ch[yy * WW + xx];
            }
        }
        float v = s * (1.0f / 9.0f) + gval;
        vals[c] = v;
        vmax = fmaxf(vmax, v);
    }
    float denom = 0.f;
    #pragma unroll
    for (int c = 0; c < NMARK; ++c) { vals[c] = __expf(vals[c] - vmax); denom += vals[c]; }
    float inv = 1.0f / denom;
    #pragma unroll
    for (int c = 0; c < NMARK; ++c)
        dst[base + (long long)c * HWSZ + rem] = (_Float16)(vals[c] * inv);
}

// ---------------------------------------------------------------------------
// Stage 7: final argmax over channels -> float index
// ---------------------------------------------------------------------------
__global__ void lw_final_argmax(const _Float16* __restrict__ lab,
                                float* __restrict__ out) {
    long long p = (long long)blockIdx.x * blockDim.x + threadIdx.x;
    if (p >= (long long)BATCH * HWSZ) return;
    int b = (int)(p / HWSZ);
    int rem = (int)(p % HWSZ);
    long long base = (long long)b * NMARK * HWSZ + rem;
    float best = -1e30f; int bi = 0;
    #pragma unroll
    for (int c = 0; c < NMARK; ++c) {
        float v = (float)lab[base + (long long)c * HWSZ];
        if (v > best) { best = v; bi = c; }
    }
    out[p] = (float)bi;
}

// ---------------------------------------------------------------------------
extern "C" void kernel_launch(void* const* d_in, const int* in_sizes, int n_in,
                              void* d_out, int out_size, void* d_ws, size_t ws_size,
                              hipStream_t stream) {
    const float* img = (const float*)d_in[0];
    const float* Kx  = (const float*)d_in[1];
    const float* Ky  = (const float*)d_in[2];
    const float* w1  = (const float*)d_in[3];
    const float* b1  = (const float*)d_in[4];
    const float* w2  = (const float*)d_in[5];
    const float* b2  = (const float*)d_in[6];
    const float* w3  = (const float*)d_in[7];
    const float* b3  = (const float*)d_in[8];
    float* out = (float*)d_out;

    const long long N = (long long)BATCH * HWSZ;

    // workspace layout
    float*    gray = (float*)d_ws;                  //  8 MB
    float*    grad = gray + N;                      //  8 MB
    _Float16* x1   = (_Float16*)(grad + N);         // 64 MB  (16 ch)
    _Float16* x2   = x1 + 16 * N;                   // 128 MB (32 ch)
    _Float16* labA = x2 + 32 * N;                   // 128 MB
    _Float16* labB = labA + 32 * N;                 // 128 MB

    const int TPB = 256;
    const int PIX_BLOCKS = (int)((N + TPB - 1) / TPB);
    const int GEMM_BLOCKS = BATCH * HH * (WW / 16) / 2;   // 2 waves per block

    lw_gray_grad<<<PIX_BLOCKS, TPB, 0, stream>>>(img, Kx, Ky, gray, grad);
    lw_conv1<<<PIX_BLOCKS, TPB, 0, stream>>>(gray, w1, b1, x1);
    lw_conv_wmma<16, true><<<GEMM_BLOCKS, 64, 0, stream>>>(x1, w2, b2, x2);
    lw_conv_wmma<32, false><<<GEMM_BLOCKS, 64, 0, stream>>>(x2, w3, b3, labA);
    lw_argmax_onehot<<<PIX_BLOCKS, TPB, 0, stream>>>(labA, labB);

    _Float16* src = labB;
    _Float16* dst = labA;
    for (int it = 0; it < NITER; ++it) {
        lw_step<<<PIX_BLOCKS, TPB, 0, stream>>>(src, grad, dst);
        _Float16* t = src; src = dst; dst = t;
    }
    // after 10 steps result sits in `src`
    lw_final_argmax<<<PIX_BLOCKS, TPB, 0, stream>>>(src, out);
}
